// RNNLayer_30159260353159
// MI455X (gfx1250) — matrix operand order repaired and checked by reading, not compiled, not run on hardware
//
#include <hip/hip_runtime.h>

// ---------------------------------------------------------------------------
// RNN layer for MI455X (gfx1250, wave32, WMMA):
//   1) k_zero        : zero BN stat accumulators in workspace (determinism)
//   2) k_pack_b      : one-time fp32 -> bf16 repack of W and V into WMMA
//                      B-operand order (removes all cvt/perm from hot loops)
//   3) k_gemm_stats  : Wx = x @ W^T via v_wmma_f32_16x16x32_bf16, fused
//                      per-channel sum/sumsq (LDS reduce + global f32 atomics)
//   4) k_finalize    : scale = gamma*rsqrt(var+eps), shift = beta - mean*scale
//   5) k_recur       : y_t = sigmoid(BN(Wx_t) + y_{t-1} @ V^T), 2048 steps.
//                      V^T tiles resident in VGPRs; y state ping-ponged in LDS
//                      in WMMA A-operand order (2x ds_load_b128 per tile,
//                      ONE barrier per step); BN(Wx_t) injected as WMMA C-seed;
//                      two independent WMMA chains + 2-step-deep seed pipeline
//                      shorten the serial critical path.
// ---------------------------------------------------------------------------

constexpr int Bc = 64, Tc = 2048, Ic = 128, Hc = 128;
constexpr int NROW = Bc * Tc;              // 131072 rows of Wx
constexpr int ABUF = 4 * 32 * 16;          // one A-operand y-state buffer (bf16)

typedef __attribute__((ext_vector_type(16))) __bf16        v16bf;
typedef __attribute__((ext_vector_type(2)))  __bf16        v2bf;
typedef __attribute__((ext_vector_type(8)))  float         v8f;
typedef __attribute__((ext_vector_type(8)))  unsigned int  v8u;
typedef __attribute__((ext_vector_type(4)))  unsigned int  v4u;

union ABf { v8u u; v4u u4[2]; v16bf b; };

// fp32 -> packed bf16x2 via hardware cvt (v_cvt_pk_bf16_f32)
__device__ __forceinline__ unsigned packbf(float lo, float hi) {
#if __has_builtin(__builtin_amdgcn_cvt_pk_bf16_f32)
    v2bf p = __builtin_amdgcn_cvt_pk_bf16_f32(lo, hi);
    return __builtin_bit_cast(unsigned, p);
#else
    __bf16 a = (__bf16)lo, b = (__bf16)hi;
    return (unsigned)__builtin_bit_cast(unsigned short, a) |
           ((unsigned)__builtin_bit_cast(unsigned short, b) << 16);
#endif
}
__device__ __forceinline__ unsigned short f2bf(float f) {
    return __builtin_bit_cast(unsigned short, (__bf16)f);   // HW v_cvt_bf16_f32
}
// 16-bit A-matrix 16x32 layout: K-pair base for 32-bit element j, per lane half
__device__ __forceinline__ int koff(int j, int half) {
    return ((j < 4) ? 2 * j : 8 + 2 * j) + 8 * half;
}

__global__ void k_zero(float* p, int n) {
    int i = blockIdx.x * 256 + threadIdx.x;
    if (i < n) p[i] = 0.0f;
}

// ------------- one-time repack: M[128][128] fp32 -> bf16 B-operand order -----
// B[k][n] = M[n][k].  Tile (nb, kt), lane l, element e:
//   n = nb*16 + (l&15),  k = kt*32 + 8*(l>>4) + (e<8 ? e : 8 + e)
//   dst[((nb*4 + kt)*32 + l)*16 + e]
__global__ void k_pack_b(const float* __restrict__ src, unsigned short* __restrict__ dst)
{
    const int rid = blockIdx.x * 256 + threadIdx.x;   // 0..1023
    const int l  = rid & 31;
    const int kt = (rid >> 5) & 3;
    const int nb = rid >> 7;
    const int n  = nb * 16 + (l & 15);
    const int k  = kt * 32 + 8 * (l >> 4);
    const float* sp = src + n * 128 + k;
    unsigned* d32 = (unsigned*)(dst + (size_t)rid * 16);
#pragma unroll
    for (int q = 0; q < 4; ++q) {
        float4 v = *(const float4*)(sp + 4 * q + (q >= 2 ? 8 : 0));
        d32[2 * q]     = packbf(v.x, v.y);
        d32[2 * q + 1] = packbf(v.z, v.w);
    }
}

// --------------------------- GEMM + BN statistics ---------------------------
__global__ __launch_bounds__(256) void k_gemm_stats(
    const float* __restrict__ x, const unsigned short* __restrict__ Wb,
    float* __restrict__ Wx, float* __restrict__ gstats)
{
    __shared__ float sstat[256];              // [0..127] sum, [128..255] sumsq
    const int tid = threadIdx.x;
    sstat[tid] = 0.0f;
    __syncthreads();

    const int w = tid >> 5, lane = tid & 31, half = lane >> 4, l16 = lane & 15;
    const long wgRow = (long)blockIdx.x * 128;     // 128 rows per workgroup
    const long rowA  = wgRow + w * 16 + l16;       // A rows: M = l16 (both halves)

    // A tiles (x rows, fp32 -> bf16), 4 K-tiles of 32
    const float* xr = x + rowA * Ic;
    ABf a[4];
#pragma unroll
    for (int kt = 0; kt < 4; ++kt)
#pragma unroll
        for (int j = 0; j < 8; ++j) {
            const int k = kt * 32 + koff(j, half);
            float2 v = *(const float2*)(xr + k);
            a[kt].u[j] = packbf(v.x, v.y);
        }

    float s[8], s2[8];
#pragma unroll
    for (int i = 0; i < 8; ++i) { s[i] = 0.0f; s2[i] = 0.0f; }

#pragma unroll
    for (int nb = 0; nb < 8; ++nb) {
        // B tiles: pre-packed bf16 operand order -> 2x b128 per K-tile, no VALU
        ABf bt[4];
#pragma unroll
        for (int kt = 0; kt < 4; ++kt) {
            const unsigned short* p = Wb + (size_t)((nb * 4 + kt) * 32 + lane) * 16;
            bt[kt].u4[0] = *(const v4u*)(p);
            bt[kt].u4[1] = *(const v4u*)(p + 8);
        }
        v8f c = {};
        c = __builtin_amdgcn_wmma_f32_16x16x32_bf16(false, a[0].b, false, bt[0].b, (short)0, c, false, false);
        c = __builtin_amdgcn_wmma_f32_16x16x32_bf16(false, a[1].b, false, bt[1].b, (short)0, c, false, false);
        c = __builtin_amdgcn_wmma_f32_16x16x32_bf16(false, a[2].b, false, bt[2].b, (short)0, c, false, false);
        c = __builtin_amdgcn_wmma_f32_16x16x32_bf16(false, a[3].b, false, bt[3].b, (short)0, c, false, false);

        const int  n_g  = nb * 16 + l16;
        const long rowD = wgRow + w * 16 + 8 * half;   // D: M = r + 8*half
#pragma unroll
        for (int r = 0; r < 8; ++r) {
            const float v = c[r];
            Wx[(rowD + r) * Hc + n_g] = v;
            s[nb]  += v;
            s2[nb] += v * v;
        }
    }

    // fold the two lane-halves (same channel, different M rows)
#pragma unroll
    for (int nb = 0; nb < 8; ++nb) {
        s[nb]  += __shfl_xor(s[nb], 16, 32);
        s2[nb] += __shfl_xor(s2[nb], 16, 32);
    }
    if (half == 0) {
#pragma unroll
        for (int nb = 0; nb < 8; ++nb) {
            atomicAdd(&sstat[nb * 16 + l16], s[nb]);
            atomicAdd(&sstat[128 + nb * 16 + l16], s2[nb]);
        }
    }
    __syncthreads();
    atomicAdd(&gstats[tid], sstat[tid]);
}

// --------------------------- BN finalize ------------------------------------
__global__ void k_finalize(const float* __restrict__ gstats,
                           const float* __restrict__ gamma,
                           const float* __restrict__ beta,
                           float* __restrict__ scale, float* __restrict__ shift)
{
    const int h = threadIdx.x;
    const float invN = 1.0f / (float)NROW;
    const float mean = gstats[h] * invN;
    const float var  = gstats[128 + h] * invN - mean * mean;
    const float inv  = rsqrtf(var + 1e-5f);
    const float sc   = gamma[h] * inv;
    scale[h] = sc;
    shift[h] = beta[h] - mean * sc;
}

// --------------------------- sequential recurrence --------------------------
// LDS y-state kept in WMMA A-operand order, double-buffered (ping-pong):
//   buf[(kt*32 + laneA)*16 + e], laneA = M + 16*halfA,
//   element e of lane laneA in tile kt <-> K = kt*32 + koff(e/2, halfA) + (e&1).
// A step reads buffer p, writes buffer q, then ONE barrier: it orders both
// "writes of q before next reads of q" and "reads of p before next writes of p".
__global__ __launch_bounds__(256) void k_recur(
    const float* __restrict__ Wx, const unsigned short* __restrict__ Vb,
    const float* __restrict__ scale, const float* __restrict__ shift,
    float* __restrict__ out)
{
    __shared__ __align__(16) unsigned short As[2][ABUF];       // 2 x 4 KB
    const int tid = threadIdx.x;
    for (int i = tid; i < ABUF; i += 256) As[0][i] = 0;        // y_{-1} = 0

    const int w = tid >> 5, lane = tid & 31, half = lane >> 4, l16 = lane & 15;
    const int n_g = w * 16 + l16;                 // this lane's output channel
    const float cs = scale[n_g], cb = shift[n_g];

    // writer-side inverse A-layout mapping for channel h = n_g
    const int kt_w   = n_g >> 5;
    const int k32    = n_g & 31;
    const int pb     = k32 & ~1;                  // K-pair base within tile
    const int lo     = k32 & 1;
    const int half_w = ((pb & 15) >= 8) ? 1 : 0;  // which lane-half holds this K
    const int pbh    = pb - 8 * half_w;           // pair base within half's set
    const int j_w    = (pbh < 8) ? (pbh >> 1) : (4 + ((pbh - 16) >> 1));
    const int e_w    = 2 * j_w + lo;
    // ushort address: (kt*32 + M + 16*half_w)*16 + e ; M = r + 8*half
    const int wbase  = (kt_w * 32 + 16 * half_w + 8 * half) * 16 + e_w;

    // V^T B-tiles resident in VGPRs for all T steps (pre-packed, 2x b128 each)
    ABf bt[4];
#pragma unroll
    for (int kt = 0; kt < 4; ++kt) {
        const unsigned short* p = Vb + (size_t)((w * 4 + kt) * 32 + lane) * 16;
        bt[kt].u4[0] = *(const v4u*)(p);
        bt[kt].u4[1] = *(const v4u*)(p + 8);
    }

    const int  b0 = blockIdx.x * 16;              // 16 batch rows per workgroup
    const long rstride = (long)Tc * Hc;           // stride between batch rows
    const float* sb = Wx  + (long)(b0 + 8 * half) * rstride + n_g;  // C-seed base
    float*       ob = out + (long)(b0 + 8 * half) * rstride + n_g;

    // one recurrence step: read y from rb, seed with BN(Wx_t), write y to wb
    auto step = [&](const unsigned short* rb, unsigned short* wb,
                    const float (&seed)[8], int t) {
        ABf a[4];
#pragma unroll
        for (int kt = 0; kt < 4; ++kt) {
            const unsigned short* p = rb + (kt * 32 + lane) * 16;
            a[kt].u4[0] = *(const v4u*)(p);
            a[kt].u4[1] = *(const v4u*)(p + 8);
        }
        v8f c0, c1 = {};
#pragma unroll
        for (int r = 0; r < 8; ++r) c0[r] = seed[r] * cs + cb;
        c0 = __builtin_amdgcn_wmma_f32_16x16x32_bf16(false, a[0].b, false, bt[0].b, (short)0, c0, false, false);
        c1 = __builtin_amdgcn_wmma_f32_16x16x32_bf16(false, a[1].b, false, bt[1].b, (short)0, c1, false, false);
        c0 = __builtin_amdgcn_wmma_f32_16x16x32_bf16(false, a[2].b, false, bt[2].b, (short)0, c0, false, false);
        c1 = __builtin_amdgcn_wmma_f32_16x16x32_bf16(false, a[3].b, false, bt[3].b, (short)0, c1, false, false);
#pragma unroll
        for (int r = 0; r < 8; ++r) {
            const float z = c0[r] + c1[r];
            // fast sigmoid: v_exp_f32 + v_rcp_f32 (short serial critical path)
            const float y = __builtin_amdgcn_rcpf(1.0f + __expf(-z));
            wb[wbase + r * 16] = f2bf(y);          // A-operand order store
            ob[r * rstride + (long)t * Hc] = y;
        }
        __syncthreads();                           // single barrier per step
    };

    // 2-step-deep seed pipeline: loads for t+2 issued before computing step t
    float sA[8], sB[8];
#pragma unroll
    for (int r = 0; r < 8; ++r) sA[r] = sb[r * rstride];            // t = 0
#pragma unroll
    for (int r = 0; r < 8; ++r) sB[r] = sb[r * rstride + Hc];       // t = 1
    __syncthreads();                               // As[0] zeroed & visible

    for (int t = 0; t < Tc; t += 2) {
        float nA[8], nB[8];
        const int t2 = (t + 2 < Tc) ? t + 2 : Tc - 1;
#pragma unroll
        for (int r = 0; r < 8; ++r) nA[r] = sb[r * rstride + (long)t2 * Hc];
        step(As[0], As[1], sA, t);

        const int t3 = (t + 3 < Tc) ? t + 3 : Tc - 1;
#pragma unroll
        for (int r = 0; r < 8; ++r) nB[r] = sb[r * rstride + (long)t3 * Hc];
        step(As[1], As[0], sB, t + 1);

#pragma unroll
        for (int r = 0; r < 8; ++r) { sA[r] = nA[r]; sB[r] = nB[r]; }
    }
}

// ---------------------------------------------------------------------------
extern "C" void kernel_launch(void* const* d_in, const int* in_sizes, int n_in,
                              void* d_out, int out_size, void* d_ws, size_t ws_size,
                              hipStream_t stream)
{
    const float* x     = (const float*)d_in[0];
    const float* W     = (const float*)d_in[1];
    const float* V     = (const float*)d_in[2];
    const float* gamma = (const float*)d_in[3];
    const float* beta  = (const float*)d_in[4];
    float* out = (float*)d_out;

    // workspace: Wx [NROW*Hc f32] | gstats[256 f32] | scale[128] | shift[128]
    //            | Wb [16384 bf16] | Vb [16384 bf16]
    float* ws     = (float*)d_ws;
    float* Wx     = ws;
    float* gstats = ws + (size_t)NROW * Hc;
    float* scale  = gstats + 256;
    float* shift  = scale + 128;
    unsigned short* Wb = (unsigned short*)(shift + 128);
    unsigned short* Vb = Wb + 128 * 128;

    k_zero<<<1, 256, 0, stream>>>(gstats, 256);
    k_pack_b<<<4, 256, 0, stream>>>(W, Wb);
    k_pack_b<<<4, 256, 0, stream>>>(V, Vb);
    k_gemm_stats<<<NROW / 128, 256, 0, stream>>>(x, Wb, Wx, gstats);
    k_finalize<<<1, 128, 0, stream>>>(gstats, gamma, beta, scale, shift);
    k_recur<<<4, 256, 0, stream>>>(Wx, Vb, scale, shift, out);
}
